// Self_Attention_21311627722866
// MI455X (gfx1250) — matrix operand order repaired and checked by reading, not compiled, not run on hardware
//
#include <hip/hip_runtime.h>
#include <cstdint>
#include <cstddef>

// ---------------------------------------------------------------------------
// Types for WMMA fragments (gfx1250 wave32: v_wmma_f32_16x16x32_bf16)
// ---------------------------------------------------------------------------
typedef __attribute__((ext_vector_type(16))) __bf16 v16bf;
typedef __attribute__((ext_vector_type(8)))  float  v8f;
typedef __attribute__((ext_vector_type(4)))  unsigned int u32x4;

union FragU { u32x4 q[2]; v16bf v; };

#define V8F_ZERO {0.f,0.f,0.f,0.f,0.f,0.f,0.f,0.f}

static __device__ __forceinline__ unsigned short f32_to_bf16(float f) {
  union { float f; unsigned int u; } c; c.f = f;
  unsigned int u = c.u;
  unsigned int r = u + 0x7FFFu + ((u >> 16) & 1u);  // round to nearest even
  return (unsigned short)(r >> 16);
}

// ---------------------------------------------------------------------------
// Problem constants
// ---------------------------------------------------------------------------
#define BATCH 8
#define SEQ   2048
#define DIM   768
#define NROW  (BATCH*SEQ)          // 16384
#define ATT_SCALE 0.125f           // 1/sqrt(64), hardcoded in reference

// ---------------------------------------------------------------------------
// f32 -> bf16 elementwise convert
// ---------------------------------------------------------------------------
__global__ void __launch_bounds__(256)
cvt_f32_bf16(const float* __restrict__ in, unsigned short* __restrict__ out, int n) {
  int i = blockIdx.x * 256 + threadIdx.x;
  if (i < n) out[i] = f32_to_bf16(in[i]);
}

// ---------------------------------------------------------------------------
// QKV projection GEMM:  [NROW x DIM] (bf16) @ [DIM x DIM] (bf16) -> bf16
// grid: (NROW/128, DIM/128, 3), block 256 (8 waves)
// ---------------------------------------------------------------------------
__global__ void __launch_bounds__(256)
qkv_gemm(const unsigned short* __restrict__ xb,
         const unsigned short* __restrict__ wb,
         unsigned short* __restrict__ qg,
         unsigned short* __restrict__ kg,
         unsigned short* __restrict__ vg) {
  __shared__ unsigned short sA[128 * 32];   // [row][d]   row-major
  __shared__ unsigned short sB[128 * 32];   // [o][d]     (transposed W chunk)

  const int t    = threadIdx.x;
  const int lane = t & 31;
  const int wave = t >> 5;
  const int lr   = lane & 15;
  const int lh   = lane >> 4;               // 0/1 half of wave

  const int rowBase = blockIdx.x * 128;
  const int colBase = blockIdx.y * 128;
  const int mat     = blockIdx.z;
  const unsigned short* W = wb + (size_t)mat * DIM * DIM;
  unsigned short* outp = (mat == 0) ? qg : ((mat == 1) ? kg : vg);

  const int wr = (wave >> 1) * 32;          // wave's rows inside 128-tile
  const int wc = (wave & 1) * 64;           // wave's cols inside 128-tile

  v8f acc[2][4];
#pragma unroll
  for (int ti = 0; ti < 2; ++ti)
#pragma unroll
    for (int n = 0; n < 4; ++n) { v8f z = V8F_ZERO; acc[ti][n] = z; }

  for (int dc = 0; dc < DIM; dc += 32) {
    __syncthreads();
    // stage A chunk: 128 rows x 32 d  (512 x uint4, 2 per thread)
#pragma unroll
    for (int i = 0; i < 2; ++i) {
      int idx = t + i * 256;
      int r = idx >> 2, c = idx & 3;
      *(u32x4*)(&sA[r * 32 + c * 8]) =
          *(const u32x4*)(xb + (size_t)(rowBase + r) * DIM + dc + c * 8);
    }
    // stage B chunk transposed: sB[o][d]  (2048 x uint, 8 per thread)
#pragma unroll
    for (int i = 0; i < 8; ++i) {
      int idx = t + i * 256;                 // 0..2047
      int d = idx >> 6;                      // 0..31
      int op = idx & 63;                     // pair index along o
      unsigned int u =
          *(const unsigned int*)(W + (size_t)(dc + d) * DIM + colBase + op * 2);
      sB[(op * 2    ) * 32 + d] = (unsigned short)(u & 0xffffu);
      sB[(op * 2 + 1) * 32 + d] = (unsigned short)(u >> 16);
    }
    __syncthreads();

#pragma unroll
    for (int ti = 0; ti < 2; ++ti) {
      FragU a;
      const int arow = wr + ti * 16 + lr;
      a.q[0] = *(const u32x4*)(&sA[arow * 32 + lh * 8]);
      a.q[1] = *(const u32x4*)(&sA[arow * 32 + lh * 8 + 16]);
#pragma unroll
      for (int n = 0; n < 4; ++n) {
        FragU b;
        const int bcol = wc + n * 16 + lr;
        b.q[0] = *(const u32x4*)(&sB[bcol * 32 + lh * 8]);
        b.q[1] = *(const u32x4*)(&sB[bcol * 32 + lh * 8 + 16]);
        acc[ti][n] = __builtin_amdgcn_wmma_f32_16x16x32_bf16(
            false, a.v, false, b.v, (short)0, acc[ti][n], false, false);
      }
    }
  }

#pragma unroll
  for (int ti = 0; ti < 2; ++ti)
#pragma unroll
    for (int n = 0; n < 4; ++n)
#pragma unroll
      for (int v = 0; v < 8; ++v) {
        int grow = rowBase + wr + ti * 16 + v + 8 * lh;
        int gcol = colBase + wc + n * 16 + lr;
        outp[(size_t)grow * DIM + gcol] = f32_to_bf16(acc[ti][n][v]);
      }
}

// ---------------------------------------------------------------------------
// Flash attention: one block = (batch b, 64-query tile).
// 8 waves: wave = (q-subtile of 16 rows) x (column half of 384).
// k-loop of 32 keys: S via 48 WMMAs, online softmax, P@V via 24 WMMAs.
// Dynamic LDS: sQ 96KB + sK 48KB + sV 48KB + sP 8KB = 200KB (< 320KB/WGP).
// ---------------------------------------------------------------------------
__global__ void __launch_bounds__(256)
attn(const unsigned short* __restrict__ qg,
     const unsigned short* __restrict__ kg,
     const unsigned short* __restrict__ vg,
     float* __restrict__ out) {
  extern __shared__ unsigned short smem[];
  unsigned short* sQ = smem;                 // [64][768]
  unsigned short* sK = sQ + 64 * DIM;        // [32 keys][768]
  unsigned short* sV = sK + 32 * DIM;        // [768 o][32 keys] (transposed)
  unsigned short* sP = sV + DIM * 32;        // 8 waves * [16][32]

  const int t    = threadIdx.x;
  const int lane = t & 31;
  const int wave = t >> 5;
  const int lr   = lane & 15;
  const int lh   = lane >> 4;

  const int b     = blockIdx.y;
  const int qtile = blockIdx.x;              // 64 rows
  const int qs    = wave >> 1;               // 0..3 (16-row subtile)
  const int ch    = wave & 1;                // 0/1 (384-col half)
  const size_t rowbase = (size_t)b * SEQ + (size_t)qtile * 64;

  // stage Q tile once: 64x768 bf16 = 6144 uint4, 24 per thread
#pragma unroll
  for (int i = 0; i < 24; ++i) {
    int idx = t + i * 256;
    int r = idx / 96, c = idx % 96;
    *(u32x4*)(&sQ[r * DIM + c * 8]) =
        *(const u32x4*)(qg + (rowbase + r) * DIM + c * 8);
  }

  v8f acc[24];
#pragma unroll
  for (int n = 0; n < 24; ++n) { v8f z = V8F_ZERO; acc[n] = z; }
  float mrow[8], lrow[8];
#pragma unroll
  for (int v = 0; v < 8; ++v) { mrow[v] = -3.0e38f; lrow[v] = 0.0f; }

  unsigned short* myP = sP + wave * 16 * 32;
  const int arow = qs * 16 + lr;

  for (int j = 0; j < SEQ; j += 32) {
    __syncthreads();
    // stage K chunk: 32x768 bf16 = 3072 uint4, 12 per thread
#pragma unroll
    for (int i = 0; i < 12; ++i) {
      int idx = t + i * 256;
      int r = idx / 96, c = idx % 96;
      *(u32x4*)(&sK[r * DIM + c * 8]) =
          *(const u32x4*)(kg + ((size_t)b * SEQ + j + r) * DIM + c * 8);
    }
    // stage V chunk transposed [o][key]: 12288 uints, 48 per thread
#pragma unroll
    for (int i = 0; i < 48; ++i) {
      int idx = t + i * 256;
      int key = idx / 384, op = idx % 384;
      unsigned int u =
          *(const unsigned int*)(vg + ((size_t)b * SEQ + j + key) * DIM + op * 2);
      sV[(op * 2    ) * 32 + key] = (unsigned short)(u & 0xffffu);
      sV[(op * 2 + 1) * 32 + key] = (unsigned short)(u >> 16);
    }
    __syncthreads();

    // ---- S = Q * K^T for 16 q-rows x 32 keys --------------------------
    v8f s0 = V8F_ZERO, s1 = V8F_ZERO;
    for (int dc = 0; dc < DIM; dc += 32) {
      FragU a, b0, b1;
      a.q[0]  = *(const u32x4*)(&sQ[arow * DIM + dc + lh * 8]);
      a.q[1]  = *(const u32x4*)(&sQ[arow * DIM + dc + lh * 8 + 16]);
      b0.q[0] = *(const u32x4*)(&sK[lr * DIM + dc + lh * 8]);
      b0.q[1] = *(const u32x4*)(&sK[lr * DIM + dc + lh * 8 + 16]);
      b1.q[0] = *(const u32x4*)(&sK[(16 + lr) * DIM + dc + lh * 8]);
      b1.q[1] = *(const u32x4*)(&sK[(16 + lr) * DIM + dc + lh * 8 + 16]);
      s0 = __builtin_amdgcn_wmma_f32_16x16x32_bf16(
          false, a.v, false, b0.v, (short)0, s0, false, false);
      s1 = __builtin_amdgcn_wmma_f32_16x16x32_bf16(
          false, a.v, false, b1.v, (short)0, s1, false, false);
    }

    // ---- online softmax ----------------------------------------------
    float alpha[8], mnew[8];
#pragma unroll
    for (int v = 0; v < 8; ++v) {
      s0[v] *= ATT_SCALE;
      s1[v] *= ATT_SCALE;
      float mx = fmaxf(s0[v], s1[v]);
      mx = fmaxf(mx, __shfl_xor(mx, 1));
      mx = fmaxf(mx, __shfl_xor(mx, 2));
      mx = fmaxf(mx, __shfl_xor(mx, 4));
      mx = fmaxf(mx, __shfl_xor(mx, 8));
      mnew[v]  = fmaxf(mrow[v], mx);
      alpha[v] = __expf(mrow[v] - mnew[v]);
      mrow[v]  = mnew[v];
    }
#pragma unroll
    for (int v = 0; v < 8; ++v) {
      float p0 = __expf(s0[v] - mnew[v]);
      float p1 = __expf(s1[v] - mnew[v]);
      float rs = p0 + p1;
      rs += __shfl_xor(rs, 1);
      rs += __shfl_xor(rs, 2);
      rs += __shfl_xor(rs, 4);
      rs += __shfl_xor(rs, 8);
      lrow[v] = lrow[v] * alpha[v] + rs;
      int r = v + 8 * lh;
      myP[r * 32 + lr]      = f32_to_bf16(p0);
      myP[r * 32 + 16 + lr] = f32_to_bf16(p1);
    }
#pragma unroll
    for (int n = 0; n < 24; ++n)
#pragma unroll
      for (int v = 0; v < 8; ++v) acc[n][v] *= alpha[v];

    // ---- O += P @ V ----------------------------------------------------
    FragU pa;
    pa.q[0] = *(const u32x4*)(&myP[lr * 32 + lh * 8]);
    pa.q[1] = *(const u32x4*)(&myP[lr * 32 + lh * 8 + 16]);
#pragma unroll
    for (int n = 0; n < 24; ++n) {
      FragU bv;
      int o = ch * 384 + n * 16 + lr;
      bv.q[0] = *(const u32x4*)(&sV[o * 32 + lh * 8]);
      bv.q[1] = *(const u32x4*)(&sV[o * 32 + lh * 8 + 16]);
      acc[n] = __builtin_amdgcn_wmma_f32_16x16x32_bf16(
          false, pa.v, false, bv.v, (short)0, acc[n], false, false);
    }
  }

  // ---- epilogue: divide by row sums, write f32 -------------------------
  float inv[8];
#pragma unroll
  for (int v = 0; v < 8; ++v) inv[v] = 1.0f / lrow[v];
#pragma unroll
  for (int n = 0; n < 24; ++n)
#pragma unroll
    for (int v = 0; v < 8; ++v) {
      int r = qs * 16 + v + 8 * lh;
      int o = ch * 384 + n * 16 + lr;
      out[(rowbase + r) * DIM + o] = acc[n][v] * inv[v];
    }
}

// ---------------------------------------------------------------------------
// Launcher
// ---------------------------------------------------------------------------
extern "C" void kernel_launch(void* const* d_in, const int* in_sizes, int n_in,
                              void* d_out, int out_size, void* d_ws, size_t ws_size,
                              hipStream_t stream) {
  const float* x = (const float*)d_in[0];      // [8,2048,768] f32
  const float* w = (const float*)d_in[1];      // [3,768,768]  f32
  float* out = (float*)d_out;                  // [8,2048,768] f32

  unsigned short* xb = (unsigned short*)d_ws;          // 16384*768 bf16
  unsigned short* wb = xb + (size_t)NROW * DIM;        // 3*768*768 bf16
  unsigned short* qg = wb + (size_t)3 * DIM * DIM;     // 16384*768 bf16
  unsigned short* kg = qg + (size_t)NROW * DIM;
  unsigned short* vg = kg + (size_t)NROW * DIM;

  const int nx = NROW * DIM;           // 12,582,912
  const int nw = 3 * DIM * DIM;        // 1,769,472
  cvt_f32_bf16<<<(nx + 255) / 256, 256, 0, stream>>>(x, xb, nx);
  cvt_f32_bf16<<<(nw + 255) / 256, 256, 0, stream>>>(w, wb, nw);

  qkv_gemm<<<dim3(NROW / 128, DIM / 128, 3), 256, 0, stream>>>(xb, wb, qg, kg, vg);

  // dynamic LDS: (64*768 + 32*768 + 768*32 + 8*16*32) * 2 bytes = 204800
  const size_t ldsBytes = (size_t)(64 * DIM + 32 * DIM + DIM * 32 + 8 * 16 * 32) * 2;
  attn<<<dim3(SEQ / 64, BATCH), 256, ldsBytes, stream>>>(qg, kg, vg, out);
}
